// DynamicMaskHead_26070451486955
// MI455X (gfx1250) — matrix operand order repaired and checked by reading, not compiled, not run on hardware
//
#include <hip/hip_runtime.h>

typedef __attribute__((ext_vector_type(16))) _Float16 v16h;
typedef __attribute__((ext_vector_type(8)))  float    v8f;

// Problem geometry (fixed by setup_inputs in the reference)
#define C_IN   8
#define H_IN   128
#define W_IN   192
#define HW_IN  (H_IN * W_IN)      // 24576
#define H_OUT  (2 * H_IN)         // 256
#define W_OUT  (2 * W_IN)         // 384
#define P_LEN  169

// ---------------------------------------------------------------------------
// Kernel 1: dynamic mask head, 2x v_wmma_f32_16x16x32_f16 per 16-pixel tile.
//   D = A x B :  A = per-instance weights (M=outch, K=inch padded to 32)
//                B = activations        (K=inch,  N=16 pixels, 1 pixel/lane)
//   D layout => lane p (<16) holds pixel p's 8 output channels in VGPRs 0..7,
//   so bias/ReLU/repack/final-dot are all in-lane (no LDS transpose).
// ---------------------------------------------------------------------------
__global__ __launch_bounds__(256) void mask_head_wmma(
    const float* __restrict__ feats,     // (N, 8, H, W)
    const float* __restrict__ params,    // (n_inst, 169)
    const float* __restrict__ iloc,      // (n_inst, 2)
    const float* __restrict__ soi,       // (n_inst,)
    const int*   __restrict__ im_inds,   // (n_inst,)
    const int*   __restrict__ stride_p,  // (1,)
    float*       __restrict__ logits)    // (n_inst, H, W) -> workspace
{
  const int lane = threadIdx.x & 31;
  const int wid  = threadIdx.x >> 5;
  const int inst = blockIdx.y;
  const int gw   = blockIdx.x * (blockDim.x >> 5) + wid;  // global wave id in x
  const int nw   = gridDim.x * (blockDim.x >> 5);

  const float* p = params + (size_t)inst * P_LEN;

  // Per-instance wave-uniform scalars
  const float ix      = iloc[inst * 2 + 0];
  const float iy      = iloc[inst * 2 + 1];
  const float inv_soi = 1.0f / soi[inst];
  const int   s       = stride_p[0];
  const float half_s  = (float)(s >> 1);
  const int   img     = im_inds[inst];
  const float* f      = feats + (size_t)img * C_IN * HW_IN;

  float b0v[8], b1v[8], w2v[8];
#pragma unroll
  for (int r = 0; r < 8; ++r) {
    b0v[r] = p[152 + r];
    b1v[r] = p[160 + r];
    w2v[r] = p[144 + r];
  }
  const float b2s = p[168];

  // Build A matrices once (contiguous k order, matching B):
  //   lanes 0-15 : M = lane, halves 0..15 = K 0..15
  //   lanes 16-31: M = lane-16, halves 0..15 = K 16..31
  // Only M = 0..7 (8 output channels) and K < 10 (layer0) / K < 8 (layer1) used.
  v16h A0 = {};   // w0: 8x10  (params[0:80])
  v16h A1 = {};   // w1: 8x8   (params[80:144])
  {
    const int m = lane & 15;
    if (lane < 8) {
#pragma unroll
      for (int j = 0; j < 8; ++j) {
        A0[j] = (_Float16)p[m * 10 + j];        // K = 0..7
        A1[j] = (_Float16)p[80 + m * 8 + j];    // K = 0..7
      }
      A0[8] = (_Float16)p[m * 10 + 8];          // K = 8
      A0[9] = (_Float16)p[m * 10 + 9];          // K = 9
    }
  }

  const int ntiles = HW_IN / 16;  // W_IN % 16 == 0 -> tiles never cross a row
  for (int t = gw; t < ntiles; t += nw) {
    const int pix0 = t * 16;
    const int y    = pix0 / W_IN;          // wave-uniform
    const int x0   = pix0 - y * W_IN;
    const int m    = lane & 15;
    const int pix  = pix0 + m;

    // B (32x16, 16-bit): lane p<16 = column p; VGPR j holds K=2j,2j+1.
    // Channels: 0,1 = rel coords; 2..9 = mask feats; K=10..31 zero.
    v16h B = {};
    if (lane < 16) {
      const float lx = (float)((x0 + m) * s) + half_s;
      const float ly = (float)(y * s) + half_s;
      B[0] = (_Float16)((ix - lx) * inv_soi);
      B[1] = (_Float16)((iy - ly) * inv_soi);
#pragma unroll
      for (int c = 0; c < 8; ++c)
        B[2 + c] = (_Float16)f[c * HW_IN + pix];
    }

    // Layer 0: one WMMA covers K=10 (padded to 32)
    v8f acc = {};
    acc = __builtin_amdgcn_wmma_f32_16x16x32_f16(
        false, A0, false, B, (short)0, acc, false, false);

    // bias + ReLU + repack into next-layer B (in-lane f32->f16)
    v16h B1 = {};
    if (lane < 16) {
#pragma unroll
      for (int r = 0; r < 8; ++r) {
        float v = acc[r] + b0v[r];
        B1[r] = (_Float16)(v > 0.0f ? v : 0.0f);
      }
    }

    // Layer 1: one WMMA covers K=8
    v8f acc2 = {};
    acc2 = __builtin_amdgcn_wmma_f32_16x16x32_f16(
        false, A1, false, B1, (short)0, acc2, false, false);

    // bias + ReLU + layer2 (1x8 dot) + bias, all in-lane
    if (lane < 16) {
      float o = b2s;
#pragma unroll
      for (int r = 0; r < 8; ++r) {
        float v = acc2[r] + b1v[r];
        o = fmaf(w2v[r], (v > 0.0f ? v : 0.0f), o);
      }
      logits[(size_t)inst * HW_IN + pix] = o;
    }
  }
}

// ---------------------------------------------------------------------------
// Kernel 2: 2x bilinear upsample, align_corners=True. Pure bandwidth:
// reads the 25MB intermediate (L2-resident), writes 100MB -> HBM-bound.
// ---------------------------------------------------------------------------
__global__ __launch_bounds__(256) void bilinear_up2(
    const float* __restrict__ in, float* __restrict__ out, int n_inst)
{
  const long long total = (long long)n_inst * H_OUT * W_OUT;
  long long idx = (long long)blockIdx.x * blockDim.x + threadIdx.x;
  if (idx >= total) return;

  const int xo = (int)(idx % W_OUT);
  long long t  = idx / W_OUT;
  const int yo = (int)(t % H_OUT);
  const int n  = (int)(t / H_OUT);

  const float sy = (float)(H_IN - 1) / (float)(H_OUT - 1);
  const float sx = (float)(W_IN - 1) / (float)(W_OUT - 1);
  const float ys = (float)yo * sy;
  const float xs = (float)xo * sx;

  int y0 = (int)ys;                    // ys >= 0 -> trunc == floor
  int y1 = min(y0 + 1, H_IN - 1);
  const float wy = ys - (float)y0;
  int x0 = (int)xs;
  int x1 = min(x0 + 1, W_IN - 1);
  const float wx = xs - (float)x0;

  const float* b = in + (size_t)n * HW_IN;
  const float v00 = b[y0 * W_IN + x0];
  const float v10 = b[y1 * W_IN + x0];
  const float v01 = b[y0 * W_IN + x1];
  const float v11 = b[y1 * W_IN + x1];

  // Same order as reference: interpolate rows in y first, then x.
  const float r0 = v00 * (1.0f - wy) + v10 * wy;
  const float r1 = v01 * (1.0f - wy) + v11 * wy;
  out[idx] = r0 * (1.0f - wx) + r1 * wx;
}

// ---------------------------------------------------------------------------
extern "C" void kernel_launch(void* const* d_in, const int* in_sizes, int n_in,
                              void* d_out, int out_size, void* d_ws, size_t ws_size,
                              hipStream_t stream) {
  const float* feats    = (const float*)d_in[0];  // mask_feats
  const float* params   = (const float*)d_in[1];  // mask_head_params
  const float* iloc     = (const float*)d_in[2];  // instance_locations
  const float* soi      = (const float*)d_in[3];  // sizes_of_interest
  const int*   im_inds  = (const int*)d_in[4];    // im_inds
  // d_in[5] = fpn_levels (unused by the math; soi already gathered)
  const int*   stride_p = (const int*)d_in[6];    // mask_feat_stride (scalar)

  const int n_inst = in_sizes[1] / P_LEN;         // 256
  float* logits = (float*)d_ws;                   // (n_inst, H_IN, W_IN) fp32

  // Head: 16 blocks/instance x 8 waves = 128 waves per instance, 12 tiles each.
  mask_head_wmma<<<dim3(16, n_inst), 256, 0, stream>>>(
      feats, params, iloc, soi, im_inds, stride_p, logits);

  const long long total = (long long)n_inst * H_OUT * W_OUT;
  const int blocks = (int)((total + 255) / 256);
  bilinear_up2<<<blocks, 256, 0, stream>>>(logits, (float*)d_out, n_inst);
}
